// GNNStack_26697516712150
// MI455X (gfx1250) — matrix operand order repaired and checked by reading, not compiled, not run on hardware
//
#include <hip/hip_runtime.h>
#include <hip/hip_bf16.h>

typedef __attribute__((ext_vector_type(16))) __bf16 v16bf;
typedef __attribute__((ext_vector_type(8)))  float  v8f;

#define HD    128
#define HEADS 4

// ======================================================================
// B pre-pack: lay B[K=128 x Nc] out in WMMA bf16 fragment order.
// Fragment (cT, f): 32 lanes x 16 bf16; lane (g,lr) element e holds
// B[f*32 + g*16 + e, cT*16 + lr].  One thread per (cT, f, lane).
// ======================================================================
__global__ void pack_B_bf16(const float* __restrict__ B, __bf16* __restrict__ Bp,
                            int Nc)
{
  int i = blockIdx.x * blockDim.x + threadIdx.x;
  if (i >= (Nc / 16) * 4 * 32) return;
  int lane = i & 31;
  int f    = (i >> 5) & 3;
  int cT   = i >> 7;
  int g = lane >> 4, lr = lane & 15;
  __bf16* dst = Bp + (size_t)i * 16;
#pragma unroll
  for (int e = 0; e < 16; ++e) {
    int k = f * 32 + g * 16 + e;
    dst[e] = (__bf16)B[(size_t)k * Nc + cT * 16 + lr];
  }
}

// ======================================================================
// WMMA GEMM, K=128 fixed:  C[M x Nc] = A[M x 128] @ B (+bias) (+Cin)
// grid = (M/16, Nc/(16*COLS)), block = 32 (one wave).
// Each wave: 16 rows x (COLS*16) cols; A fragment reused COLS times.
// No LDS, no barriers; A via b128 global loads (one contiguous 16-float
// run per lane per fragment), B via coalesced packed-bf16 loads.
// ======================================================================
template <int COLS>
__global__ void __launch_bounds__(32)
wmma_gemm_k128(const float* __restrict__ A, const __bf16* __restrict__ Bp,
               const float* __restrict__ bias, const float* __restrict__ Cin,
               float* __restrict__ C, int Nc)
{
  const int lane = threadIdx.x;
  const int g  = lane >> 4;
  const int lr = lane & 15;
  const size_t arow = (size_t)(blockIdx.x * 16 + lr) * 128;

  v8f acc[COLS];
#pragma unroll
  for (int cc = 0; cc < COLS; ++cc) acc[cc] = (v8f){};

#pragma unroll
  for (int f = 0; f < 4; ++f) {
    // A 16x32 fragment: elem e -> K=f*32+g*16+e
    const float4 a0 = *(const float4*)&A[arow + f * 32 + g * 16];
    const float4 a1 = *(const float4*)&A[arow + f * 32 + g * 16 + 4];
    const float4 a2 = *(const float4*)&A[arow + f * 32 + g * 16 + 8];
    const float4 a3 = *(const float4*)&A[arow + f * 32 + g * 16 + 12];
    v16bf a;
    a[0]  = (__bf16)a0.x; a[1]  = (__bf16)a0.y; a[2]  = (__bf16)a0.z; a[3]  = (__bf16)a0.w;
    a[4]  = (__bf16)a1.x; a[5]  = (__bf16)a1.y; a[6]  = (__bf16)a1.z; a[7]  = (__bf16)a1.w;
    a[8]  = (__bf16)a2.x; a[9]  = (__bf16)a2.y; a[10] = (__bf16)a2.z; a[11] = (__bf16)a2.w;
    a[12] = (__bf16)a3.x; a[13] = (__bf16)a3.y; a[14] = (__bf16)a3.z; a[15] = (__bf16)a3.w;

#pragma unroll
    for (int cc = 0; cc < COLS; ++cc) {
      const int cT = blockIdx.y * COLS + cc;
      const v16bf b = *(const v16bf*)&Bp[((size_t)(cT * 4 + f) * 32 + lane) * 16];
      acc[cc] = __builtin_amdgcn_wmma_f32_16x16x32_bf16(false, a, false, b,
                                                        (short)0, acc[cc],
                                                        false, false);
    }
  }

  // C layout: VGPR r -> row = g*8 + r, col = lr (within the 16x16 tile)
#pragma unroll
  for (int cc = 0; cc < COLS; ++cc) {
    const int col = (blockIdx.y * COLS + cc) * 16 + lr;
    const float bv = bias ? bias[col] : 0.0f;
#pragma unroll
    for (int r = 0; r < 8; ++r) {
      const int orow = blockIdx.x * 16 + g * 8 + r;
      float val = acc[cc][r] + bv;
      const size_t idx = (size_t)orow * Nc + col;
      if (Cin) val += Cin[idx];
      C[idx] = val;
    }
  }
}

// ======================================================================
// Helpers
// ======================================================================
__global__ void zero_f(float* __restrict__ p, int n) {
  int i = blockIdx.x * blockDim.x + threadIdx.x;
  if (i < n) p[i] = 0.0f;
}

// Block-diagonal matrices from We (32 x 128):
//   M1[h*32+d, h*32+c] = We[c, h*32+d]   (for G = q @ M1)
//   M2[h*32+c, h*32+d] = We[c, h*32+d]   (for out += P @ M2)
__global__ void build_blockdiag(const float* __restrict__ We,
                                float* __restrict__ M1, float* __restrict__ M2) {
  int i = blockIdx.x * blockDim.x + threadIdx.x;
  if (i >= HD * HD) return;
  int rowIdx = i >> 7, colIdx = i & 127;
  int hr = rowIdx >> 5, hc = colIdx >> 5;
  float m1 = 0.0f, m2 = 0.0f;
  if (hr == hc) {
    int dA = rowIdx & 31, cA = colIdx & 31;
    m1 = We[cA * HD + hr * 32 + dA];
    m2 = We[(rowIdx & 31) * HD + colIdx];
  }
  M1[i] = m1;
  M2[i] = m2;
}

// order-preserving float<->uint for atomicMax on floats
__device__ __forceinline__ unsigned f2ord(float f) {
  unsigned u = __float_as_uint(f);
  return (u & 0x80000000u) ? ~u : (u | 0x80000000u);
}
__device__ __forceinline__ float ord2f(unsigned u) {
  unsigned b = (u & 0x80000000u) ? (u & 0x7fffffffu) : ~u;
  return __uint_as_float(b);
}

// ======================================================================
// Pass A: one wave per edge; 8 lanes per head, 4 dims per lane.
//   t[e,h] = scale * ( q[dst]·(k[src]+be) + ea·G[dst] );  m_ord = segment max
// ======================================================================
__global__ void edge_alpha(const float* __restrict__ q, const float* __restrict__ k,
                           const float* __restrict__ G,  const float* __restrict__ ea,
                           const float* __restrict__ be,
                           const int* __restrict__ src, const int* __restrict__ dst,
                           float* __restrict__ t, unsigned* __restrict__ m_ord,
                           int E, float scale)
{
  int wid  = (blockIdx.x * blockDim.x + threadIdx.x) >> 5;
  int lane = threadIdx.x & 31;
  if (wid >= E) return;
  int s = src[wid], d = dst[wid];
  int h = lane >> 3, j = lane & 7;
  int dd = h * 32 + j * 4;

  float4 qv  = *(const float4*)&q [(size_t)d * HD + dd];
  float4 kv  = *(const float4*)&k [(size_t)s * HD + dd];
  float4 gv  = *(const float4*)&G [(size_t)d * HD + dd];
  float4 bev = *(const float4*)&be[dd];
  float4 eav = *(const float4*)&ea[(size_t)wid * 32 + j * 4];

  float p = qv.x * (kv.x + bev.x) + qv.y * (kv.y + bev.y) +
            qv.z * (kv.z + bev.z) + qv.w * (kv.w + bev.w) +
            eav.x * gv.x + eav.y * gv.y + eav.z * gv.z + eav.w * gv.w;
  p += __shfl_xor(p, 1, 32);
  p += __shfl_xor(p, 2, 32);
  p += __shfl_xor(p, 4, 32);
  p *= scale;
  if (j == 0) {
    t[(size_t)wid * HEADS + h] = p;
    atomicMax(&m_ord[(size_t)d * HEADS + h], f2ord(p));
  }
}

// Pass B: one thread per edge: ex = exp(t - m[dst]); t <- ex; denom += ex
__global__ void edge_expsum(const int* __restrict__ dst, float* __restrict__ t,
                            const unsigned* __restrict__ m_ord,
                            float* __restrict__ denom, int E)
{
  int e = blockIdx.x * blockDim.x + threadIdx.x;
  if (e >= E) return;
  int d = dst[e];
  float4 tv = *(const float4*)&t[(size_t)e * HEADS];
  float ex0 = __expf(tv.x - ord2f(m_ord[(size_t)d * HEADS + 0]));
  float ex1 = __expf(tv.y - ord2f(m_ord[(size_t)d * HEADS + 1]));
  float ex2 = __expf(tv.z - ord2f(m_ord[(size_t)d * HEADS + 2]));
  float ex3 = __expf(tv.w - ord2f(m_ord[(size_t)d * HEADS + 3]));
  *(float4*)&t[(size_t)e * HEADS] = make_float4(ex0, ex1, ex2, ex3);
  atomicAdd(&denom[(size_t)d * HEADS + 0], ex0);
  atomicAdd(&denom[(size_t)d * HEADS + 1], ex1);
  atomicAdd(&denom[(size_t)d * HEADS + 2], ex2);
  atomicAdd(&denom[(size_t)d * HEADS + 3], ex3);
}

// Pass C: one wave per edge.
//   alpha = ex/(denom+1e-16); agg[dst] += alpha*v[src];
//   P[dst] += alpha*ea; sumA[dst,h] += alpha
__global__ void edge_aggregate(const float* __restrict__ v, const float* __restrict__ ea,
                               const float* __restrict__ t, const float* __restrict__ denom,
                               const int* __restrict__ src, const int* __restrict__ dst,
                               float* __restrict__ agg, float* __restrict__ P,
                               float* __restrict__ sumA, int E)
{
  int wid  = (blockIdx.x * blockDim.x + threadIdx.x) >> 5;
  int lane = threadIdx.x & 31;
  if (wid >= E) return;
  int s = src[wid], d = dst[wid];
  int h = lane >> 3, j = lane & 7;

  float alpha = t[(size_t)wid * HEADS + h] /
                (denom[(size_t)d * HEADS + h] + 1e-16f);

  float4 vv  = *(const float4*)&v [(size_t)s * HD + h * 32 + j * 4];
  float4 eav = *(const float4*)&ea[(size_t)wid * 32 + j * 4];

  size_t base = (size_t)d * HD + h * 32 + j * 4;
  atomicAdd(&agg[base + 0], alpha * vv.x);
  atomicAdd(&agg[base + 1], alpha * vv.y);
  atomicAdd(&agg[base + 2], alpha * vv.z);
  atomicAdd(&agg[base + 3], alpha * vv.w);
  atomicAdd(&P[base + 0], alpha * eav.x);
  atomicAdd(&P[base + 1], alpha * eav.y);
  atomicAdd(&P[base + 2], alpha * eav.z);
  atomicAdd(&P[base + 3], alpha * eav.w);
  if (j == 0) atomicAdd(&sumA[(size_t)d * HEADS + h], alpha);
}

// finalize: h_out = relu(agg + sumA[n,h]*be[hd])
__global__ void finalize_relu(const float* __restrict__ agg, const float* __restrict__ sumA,
                              const float* __restrict__ be, float* __restrict__ hout, int N)
{
  int i = blockIdx.x * blockDim.x + threadIdx.x;
  if (i >= N * HD) return;
  int n = i >> 7, hd = i & 127, h = hd >> 5;
  float val = agg[i] + sumA[(size_t)n * HEADS + h] * be[hd];
  hout[i] = val > 0.0f ? val : 0.0f;
}

// ======================================================================
extern "C" void kernel_launch(void* const* d_in, const int* in_sizes, int n_in,
                              void* d_out, int out_size, void* d_ws, size_t ws_size,
                              hipStream_t stream)
{
  (void)n_in; (void)out_size; (void)ws_size;
  const float* x  = (const float*)d_in[0];
  const int*   ei = (const int*)  d_in[1];
  const float* ea = (const float*)d_in[2];
  const int N = in_sizes[0] / HD;   // NODE_DIM == HD == 128
  const int E = in_sizes[1] / 2;
  const int* src = ei;
  const int* dst = ei + E;
  const float* Wlin = (const float*)d_in[3 + 3 * 10 + 0];
  const float* blin = (const float*)d_in[3 + 3 * 10 + 1];

  // --- workspace carve (256B aligned) ---
  char* w = (char*)d_ws;
  auto carve = [&](size_t bytes) -> char* {
    char* p = w; w += (bytes + 255) & ~(size_t)255; return p;
  };
  float*    qb   = (float*)carve((size_t)N * HD * 4);
  float*    kbuf = (float*)carve((size_t)N * HD * 4);
  float*    vb   = (float*)carve((size_t)N * HD * 4);
  float*    agg  = (float*)carve((size_t)N * HD * 4);
  float*    Gb   = (float*)carve((size_t)N * HD * 4);
  float*    Pb   = (float*)carve((size_t)N * HD * 4);
  float*    h0   = (float*)carve((size_t)N * HD * 4);
  float*    h1   = (float*)carve((size_t)N * HD * 4);
  float*    tb   = (float*)carve((size_t)E * HEADS * 4);
  unsigned* mo   = (unsigned*)carve((size_t)N * HEADS * 4);
  float*    den  = (float*)carve((size_t)N * HEADS * 4);
  float*    sA   = (float*)carve((size_t)N * HEADS * 4);
  float*    M1   = (float*)carve((size_t)HD * HD * 4);
  float*    M2   = (float*)carve((size_t)HD * HD * 4);
  __bf16*   Bp   = (__bf16*)carve((size_t)(HD / 16) * 4 * 32 * 16 * 2);  // 32KB

  const float scale = 0.1767766953f;  // 1/sqrt(32)
  const dim3 gg(N / 16, HD / (16 * 4));   // node GEMMs: COLS=4 -> y=2
  const dim3 gl(N / 16, 1);               // final GEMM: COLS=2, Nc=32
  const int  packBlocks128 = ((HD / 16) * 128 + 255) / 256;
  const int  packBlocks32  = ((32 / 16) * 128 + 255) / 256;
  const int  nhBlocks = (N * HEADS + 255) / 256;
  const int  ndBlocks = (N * HD + 255) / 256;
  const int  ewBlocks = (E + 7) / 8;      // one wave per edge
  const int  etBlocks = (E + 255) / 256;  // one thread per edge

  const float* hin = x;
  float* hbufs[2] = { h0, h1 };

  auto gemm128 = [&](const float* A, const float* B, const float* bias,
                     const float* Cin, float* C) {
    pack_B_bf16<<<packBlocks128, 256, 0, stream>>>(B, Bp, HD);
    wmma_gemm_k128<4><<<gg, 32, 0, stream>>>(A, Bp, bias, Cin, C, HD);
  };

  for (int b = 0; b < 3; ++b) {
    const float* Wq = (const float*)d_in[3 + b * 10 + 0];
    const float* bq = (const float*)d_in[3 + b * 10 + 1];
    const float* Wk = (const float*)d_in[3 + b * 10 + 2];
    const float* bk = (const float*)d_in[3 + b * 10 + 3];
    const float* Wv = (const float*)d_in[3 + b * 10 + 4];
    const float* bv = (const float*)d_in[3 + b * 10 + 5];
    const float* We = (const float*)d_in[3 + b * 10 + 6];
    const float* be = (const float*)d_in[3 + b * 10 + 7];
    const float* Ws = (const float*)d_in[3 + b * 10 + 8];
    const float* bs = (const float*)d_in[3 + b * 10 + 9];

    // dense projections (WMMA)
    gemm128(hin, Wq, bq, nullptr, qb);
    gemm128(hin, Wk, bk, nullptr, kbuf);
    gemm128(hin, Wv, bv, nullptr, vb);
    gemm128(hin, Ws, bs, nullptr, agg);   // skip connection

    // edge-projection algebra: G = q @ blockdiag(We^T)
    build_blockdiag<<<(HD * HD + 255) / 256, 256, 0, stream>>>(We, M1, M2);
    gemm128(qb, M1, nullptr, nullptr, Gb);

    // per-block accumulators
    zero_f<<<ndBlocks, 256, 0, stream>>>(Pb, N * HD);
    zero_f<<<nhBlocks, 256, 0, stream>>>(den, N * HEADS);
    zero_f<<<nhBlocks, 256, 0, stream>>>(sA,  N * HEADS);
    zero_f<<<nhBlocks, 256, 0, stream>>>((float*)mo, N * HEADS);  // ord(0) ~ -inf

    // segment softmax + aggregation
    edge_alpha    <<<ewBlocks, 256, 0, stream>>>(qb, kbuf, Gb, ea, be, src, dst, tb, mo, E, scale);
    edge_expsum   <<<etBlocks, 256, 0, stream>>>(dst, tb, mo, den, E);
    edge_aggregate<<<ewBlocks, 256, 0, stream>>>(vb, ea, tb, den, src, dst, agg, Pb, sA, E);

    // agg += P @ blockdiag(We)   (edge-value projection part, WMMA)
    gemm128(Pb, M2, nullptr, agg, agg);

    // + sumA*be, relu
    finalize_relu<<<ndBlocks, 256, 0, stream>>>(agg, sA, be, hbufs[b & 1], N);
    hin = hbufs[b & 1];
  }

  // final linear: out = h @ Wlin + blin  (N x 128 @ 128 x 32), COLS=2
  pack_B_bf16<<<packBlocks32, 256, 0, stream>>>(Wlin, Bp, 32);
  wmma_gemm_k128<2><<<gl, 32, 0, stream>>>(hin, Bp, blin, nullptr, (float*)d_out, 32);
}
